// MPVConv_58849641890574
// MI455X (gfx1250) — matrix operand order repaired and checked by reading, not compiled, not run on hardware
//
#include <hip/hip_runtime.h>
#include <hip/hip_bf16.h>
#include <math.h>

// ---------------------------------------------------------------------------
// MPVConv for MI455X (gfx1250): implicit-GEMM 3D convs on v_wmma_f32_16x16x32_f16.
// Grids stored channel-last fp16 [B][32][32][32][64] so A-fragments are two
// contiguous 16B loads per lane. Weights pre-swizzled into WMMA B-fragment
// layout with eval-mode BN folded in. Conv kernel is templated on kernel size
// so the 27-tap loop fully unrolls; B-fragments for all 4 N-tiles are loaded
// into distinct registers before the WMMA burst (one wait, 8 WMMAs).
// ---------------------------------------------------------------------------

typedef __attribute__((ext_vector_type(16))) _Float16 v16h;
typedef __attribute__((ext_vector_type(8)))  _Float16 v8h;
typedef __attribute__((ext_vector_type(8)))  float    v8f;

#define BB 8
#define NN 16384
#define CC 64
#define RR 32
#define VV 32768                    // 32^3

__device__ __forceinline__ v16h pack16(const _Float16* p0, const _Float16* p1) {
    v8h a = *(const v8h*)p0;
    v8h b = *(const v8h*)p1;
    v16h r;
#pragma unroll
    for (int e = 0; e < 8; ++e) { r[e] = a[e]; r[e + 8] = b[e]; }
    return r;
}

// ---------------------------------------------------------------------------
// per-batch coordinate stats: mean (3) + scale = 1/(2*max||c-mean||)
// ---------------------------------------------------------------------------
__global__ void mpv_stats_kernel(const float* __restrict__ coords, float* __restrict__ stats) {
    __shared__ float red[256];
    const int b = blockIdx.x;
    const float* cb = coords + (size_t)b * 3 * NN;
    float mean[3];
    for (int k = 0; k < 3; ++k) {
        float s = 0.f;
        for (int n = threadIdx.x; n < NN; n += 256) s += cb[(size_t)k * NN + n];
        red[threadIdx.x] = s; __syncthreads();
        for (int off = 128; off > 0; off >>= 1) {
            if (threadIdx.x < off) red[threadIdx.x] += red[threadIdx.x + off];
            __syncthreads();
        }
        mean[k] = red[0] / (float)NN;
        __syncthreads();
    }
    float mx = 0.f;
    for (int n = threadIdx.x; n < NN; n += 256) {
        float dx = cb[n] - mean[0];
        float dy = cb[NN + n] - mean[1];
        float dz = cb[2 * NN + n] - mean[2];
        mx = fmaxf(mx, sqrtf(dx * dx + dy * dy + dz * dz));
    }
    red[threadIdx.x] = mx; __syncthreads();
    for (int off = 128; off > 0; off >>= 1) {
        if (threadIdx.x < off) red[threadIdx.x] = fmaxf(red[threadIdx.x], red[threadIdx.x + off]);
        __syncthreads();
    }
    if (threadIdx.x < 3) stats[b * 4 + threadIdx.x] = mean[threadIdx.x];
    if (threadIdx.x == 3) stats[b * 4 + 3] = 1.0f / (red[0] * 2.0f);
}

// nc in [0,31] per axis + flattened voxel index
__global__ void mpv_nc_idx_kernel(const float* __restrict__ coords, const float* __restrict__ stats,
                                  float* __restrict__ nc, int* __restrict__ idx) {
    int gid = blockIdx.x * blockDim.x + threadIdx.x;
    if (gid >= BB * NN) return;
    int b = gid / NN, n = gid - b * NN;
    const float* st = stats + b * 4;
    int vi[3];
#pragma unroll
    for (int k = 0; k < 3; ++k) {
        float v = (coords[((size_t)b * 3 + k) * NN + n] - st[k]) * st[3] + 0.5f;
        v = fminf(fmaxf(v * (float)RR, 0.0f), (float)(RR - 1));
        nc[((size_t)b * 3 + k) * NN + n] = v;
        vi[k] = (int)rintf(v);
    }
    idx[gid] = vi[0] * (RR * RR) + vi[1] * RR + vi[2];
}

// ---------------------------------------------------------------------------
// weight swizzle into WMMA B-fragment layout, BN folded:
// wB[((t*2+kc)*4+nt)*32+lane][16 halfs]; element e -> K=(e&7)+(e&8?16:0)+(lane>=16?8:0), N=lane&15
// ---------------------------------------------------------------------------
__global__ void mpv_prep_w_kernel(const float* __restrict__ w, const float* __restrict__ g,
                                  float bns, int ntaps, _Float16* __restrict__ wB) {
    int gid = blockIdx.x * blockDim.x + threadIdx.x;
    int total = ntaps * 2 * 4 * 32 * 16;
    if (gid >= total) return;
    int e    = gid & 15;
    int lane = (gid >> 4) & 31;
    int nt   = (gid >> 9) & 3;
    int kc   = (gid >> 11) & 1;
    int t    = gid >> 12;
    int K  = (e & 7) + ((e & 8) ? 16 : 0) + ((lane >= 16) ? 8 : 0);
    int ci = kc * 32 + K;
    int co = nt * 16 + (lane & 15);
    float val;
    if (ntaps == 27) {
        int dd = t / 9, rem = t % 9, dh = rem / 3, dw = rem % 3;
        val = w[((((size_t)co * 64 + ci) * 3 + dd) * 3 + dh) * 3 + dw];
    } else {
        val = w[(size_t)co * 64 + ci];
    }
    wB[gid] = (_Float16)(val * g[co] * bns);
}

__global__ void mpv_prep_bias_kernel(const float* __restrict__ bconv, const float* __restrict__ g,
                                     const float* __restrict__ be, float bns, float* __restrict__ bias) {
    int c = threadIdx.x;
    if (c < 64) bias[c] = bconv[c] * g[c] * bns + be[c];
}

// ---------------------------------------------------------------------------
// helpers: zero, feature layout conversion, scatter-mean voxelize
// ---------------------------------------------------------------------------
__global__ void mpv_zero_kernel(float4* __restrict__ p, int n4) {
    int i = blockIdx.x * blockDim.x + threadIdx.x;
    if (i < n4) p[i] = make_float4(0.f, 0.f, 0.f, 0.f);
}

// features [B][64][N] f32 -> fh [(B*N)][64] fp16
__global__ void mpv_feat_convert_kernel(const float* __restrict__ f, _Float16* __restrict__ fh) {
    size_t gid = (size_t)blockIdx.x * blockDim.x + threadIdx.x;
    if (gid >= (size_t)BB * CC * NN) return;
    size_t n = gid % NN;
    size_t c = (gid / NN) % CC;
    size_t b = gid / ((size_t)CC * NN);
    fh[(b * NN + n) * CC + c] = (_Float16)f[gid];
}

// scatter-add channel-last fp16 points into f32 accumulator (+ optional counts)
__global__ void mpv_scatter_kernel(const _Float16* __restrict__ fh, const int* __restrict__ idx,
                                   float* __restrict__ accum, float* __restrict__ cnt, int docnt) {
    size_t gid = (size_t)blockIdx.x * blockDim.x + threadIdx.x;
    if (gid >= (size_t)BB * NN * CC) return;
    int c = (int)(gid & 63);
    size_t pn = gid >> 6;                    // b*N + n
    int b = (int)(pn / NN);
    int v = idx[pn];
    atomicAdd(&accum[(((size_t)b * VV) + v) * CC + c], (float)fh[gid]);
    if (docnt && c == 0) atomicAdd(&cnt[(size_t)b * VV + v], 1.0f);
}

__global__ void mpv_normalize_kernel(const float* __restrict__ accum, const float* __restrict__ cnt,
                                     _Float16* __restrict__ gh) {
    size_t gid = (size_t)blockIdx.x * blockDim.x + threadIdx.x;
    if (gid >= (size_t)BB * VV * CC) return;
    size_t bv = gid >> 6;
    float inv = 1.0f / fmaxf(cnt[bv], 1.0f);
    gh[gid] = (_Float16)(accum[gid] * inv);
}

// ---------------------------------------------------------------------------
// WMMA implicit-GEMM conv3d, folded BN + leaky ReLU.
// One wave per 32-voxel (full W row) x 64-channel output tile.
// K3=1: 27 taps fully unrolled -> 432 WMMAs/wave.  K3=0: 1x1x1, 16 WMMAs/wave.
// ---------------------------------------------------------------------------
template <int K3>
__global__ void __launch_bounds__(128)
mpv_conv3d_wmma_kernel(const _Float16* __restrict__ gin, const _Float16* __restrict__ wB,
                       const float* __restrict__ bias, _Float16* __restrict__ gout) {
    const int wave = threadIdx.x >> 5;
    const int lane = threadIdx.x & 31;
    const int tile = blockIdx.x * 4 + wave;        // B*32*32 = 8192 tiles (b,d,h)
    const int h = tile & 31;
    const int d = (tile >> 5) & 31;
    const int b = tile >> 10;

    const int m  = lane & 15;
    const int hi = lane >> 4;

    v8f acc[2][4];
#pragma unroll
    for (int mt = 0; mt < 2; ++mt)
#pragma unroll
        for (int nt = 0; nt < 4; ++nt) acc[mt][nt] = (v8f){};

    const int ntaps = K3 ? 27 : 1;
#pragma unroll
    for (int t = 0; t < ntaps; ++t) {
        const int dd = K3 ? (t / 9) - 1 : 0;
        const int dh = K3 ? ((t / 3) % 3) - 1 : 0;
        const int dw = K3 ? (t % 3) - 1 : 0;
        const int d2 = d + dd, h2 = h + dh;
        if (K3 && ((unsigned)d2 >= 32u || (unsigned)h2 >= 32u)) continue;  // wave-uniform
        const int w0 = m + dw;            // voxel w for M-tile 0
        const int w1 = m + 16 + dw;       // voxel w for M-tile 1
        const bool ok0 = (unsigned)w0 < 32u;
        const bool ok1 = (unsigned)w1 < 32u;
        const _Float16* rowbase = gin + ((((size_t)b * 32 + d2) * 32 + h2) * 32) * CC;
#pragma unroll
        for (int kc = 0; kc < 2; ++kc) {
            const int co = kc * 32 + hi * 8;
            v16h af0 = {}, af1 = {};
            if (ok0) af0 = pack16(rowbase + (size_t)w0 * CC + co,
                                  rowbase + (size_t)w0 * CC + co + 16);
            if (ok1) af1 = pack16(rowbase + (size_t)w1 * CC + co,
                                  rowbase + (size_t)w1 * CC + co + 16);
            // load all 4 B-fragments into distinct registers first (one clause)
            const _Float16* wbase = wB + (((size_t)(t * 2 + kc) * 4) * 32 + lane) * 16;
            v16h bf[4];
#pragma unroll
            for (int nt = 0; nt < 4; ++nt)
                bf[nt] = pack16(wbase + (size_t)nt * 512, wbase + (size_t)nt * 512 + 8);
            // WMMA burst: 8 matrix ops, B reused across both M-tiles
#pragma unroll
            for (int nt = 0; nt < 4; ++nt) {
                acc[0][nt] = __builtin_amdgcn_wmma_f32_16x16x32_f16(
                    false, af0, false, bf[nt], (short)0, acc[0][nt], false, false);
                acc[1][nt] = __builtin_amdgcn_wmma_f32_16x16x32_f16(
                    false, af1, false, bf[nt], (short)0, acc[1][nt], false, false);
            }
        }
    }

    _Float16* obase = gout + ((((size_t)b * 32 + d) * 32 + h) * 32) * CC;
#pragma unroll
    for (int mt = 0; mt < 2; ++mt) {
#pragma unroll
        for (int nt = 0; nt < 4; ++nt) {
            const int ch = nt * 16 + (lane & 15);
            const float bv = bias[ch];
#pragma unroll
            for (int j = 0; j < 8; ++j) {
                const int mm = mt * 16 + j + hi * 8;  // C/D layout: VGPR j -> M=j (+8 hi)
                float v = acc[mt][nt][j] + bv;
                v = (v >= 0.f) ? v : 0.1f * v;        // leaky relu
                obase[(size_t)mm * CC + ch] = (_Float16)v;
            }
        }
    }
}

// ---------------------------------------------------------------------------
// WMMA point-MLP: [rows][64] x W(64x64) + folded BN, ReLU.
// One wave per 32-row x 64-channel tile -> 16 WMMAs, B reused across M-tiles.
// ---------------------------------------------------------------------------
__global__ void __launch_bounds__(128)
mpv_pointmlp_wmma_kernel(const _Float16* __restrict__ xin, const _Float16* __restrict__ wB,
                         const float* __restrict__ bias, _Float16* __restrict__ yout) {
    const int wave = threadIdx.x >> 5;
    const int lane = threadIdx.x & 31;
    const int tile = blockIdx.x * 4 + wave;        // (B*N)/32 tiles
    const int row0 = tile * 32;
    const int m  = lane & 15;
    const int hi = lane >> 4;

    v8f acc[2][4];
#pragma unroll
    for (int mt = 0; mt < 2; ++mt)
#pragma unroll
        for (int nt = 0; nt < 4; ++nt) acc[mt][nt] = (v8f){};

#pragma unroll
    for (int kc = 0; kc < 2; ++kc) {
        const int co = kc * 32 + hi * 8;
        const _Float16* a0 = xin + (size_t)(row0 + m) * CC + co;
        const _Float16* a1 = xin + (size_t)(row0 + 16 + m) * CC + co;
        v16h af0 = pack16(a0, a0 + 16);
        v16h af1 = pack16(a1, a1 + 16);
        const _Float16* wbase = wB + (((size_t)kc * 4) * 32 + lane) * 16;
        v16h bf[4];
#pragma unroll
        for (int nt = 0; nt < 4; ++nt)
            bf[nt] = pack16(wbase + (size_t)nt * 512, wbase + (size_t)nt * 512 + 8);
#pragma unroll
        for (int nt = 0; nt < 4; ++nt) {
            acc[0][nt] = __builtin_amdgcn_wmma_f32_16x16x32_f16(
                false, af0, false, bf[nt], (short)0, acc[0][nt], false, false);
            acc[1][nt] = __builtin_amdgcn_wmma_f32_16x16x32_f16(
                false, af1, false, bf[nt], (short)0, acc[1][nt], false, false);
        }
    }

#pragma unroll
    for (int mt = 0; mt < 2; ++mt) {
        _Float16* obase = yout + (size_t)(row0 + mt * 16) * CC;
#pragma unroll
        for (int nt = 0; nt < 4; ++nt) {
            const int ch = nt * 16 + (lane & 15);
            const float bv = bias[ch];
#pragma unroll
            for (int j = 0; j < 8; ++j) {
                const int mm = j + hi * 8;
                float v = acc[mt][nt][j] + bv;
                v = fmaxf(v, 0.f);                    // relu
                obase[(size_t)mm * CC + ch] = (_Float16)v;
            }
        }
    }
}

// ---------------------------------------------------------------------------
// trilinear devoxelize (+fusion variants)
// ---------------------------------------------------------------------------
__device__ __forceinline__ float mpv_devox(const _Float16* __restrict__ grid,
                                           const float* __restrict__ nc,
                                           int b, int n, int c) {
    const float x = nc[((size_t)b * 3 + 0) * NN + n];
    const float y = nc[((size_t)b * 3 + 1) * NN + n];
    const float z = nc[((size_t)b * 3 + 2) * NN + n];
    const int lx = (int)floorf(x), ly = (int)floorf(y), lz = (int)floorf(z);
    const float fx = x - lx, fy = y - ly, fz = z - lz;
    const int hx = min(lx + 1, RR - 1), hy = min(ly + 1, RR - 1), hz = min(lz + 1, RR - 1);
    const float wx[2] = {1.f - fx, fx}; const int xs[2] = {lx, hx};
    const float wy[2] = {1.f - fy, fy}; const int ys[2] = {ly, hy};
    const float wz[2] = {1.f - fz, fz}; const int zs[2] = {lz, hz};
    float s = 0.f;
#pragma unroll
    for (int i = 0; i < 2; ++i)
#pragma unroll
        for (int j = 0; j < 2; ++j)
#pragma unroll
            for (int k = 0; k < 2; ++k) {
                const int vid = xs[i] * (RR * RR) + ys[j] * RR + zs[k];
                s += wx[i] * wy[j] * wz[k] *
                     (float)grid[((size_t)b * VV + vid) * CC + c];
            }
    return s;
}

// fused1 = devox(grid) + pmlp1;  also keep v1
__global__ void mpv_devox_fuse1_kernel(const _Float16* __restrict__ grid, const float* __restrict__ nc,
                                       const _Float16* __restrict__ pm,
                                       _Float16* __restrict__ v1, _Float16* __restrict__ fused1) {
    size_t gid = (size_t)blockIdx.x * blockDim.x + threadIdx.x;
    if (gid >= (size_t)BB * NN * CC) return;
    int c = (int)(gid & 63);
    size_t pn = gid >> 6;
    int b = (int)(pn / NN), n = (int)(pn % NN);
    float s = mpv_devox(grid, nc, b, n, c);
    v1[gid] = (_Float16)s;
    fused1[gid] = (_Float16)(s + (float)pm[gid]);
}

// out[b][c][n] = devox(grid) + pmlp2 + v1   (transposed to channel-first)
__global__ void mpv_final_kernel(const _Float16* __restrict__ grid, const float* __restrict__ nc,
                                 const _Float16* __restrict__ pm, const _Float16* __restrict__ v1,
                                 float* __restrict__ out) {
    size_t gid = (size_t)blockIdx.x * blockDim.x + threadIdx.x;
    if (gid >= (size_t)BB * NN * CC) return;
    int c = (int)(gid & 63);
    size_t pn = gid >> 6;
    int b = (int)(pn / NN), n = (int)(pn % NN);
    float s = mpv_devox(grid, nc, b, n, c) + (float)pm[gid] + (float)v1[gid];
    out[((size_t)b * CC + c) * NN + n] = s;
}

__global__ void mpv_copy_coords_kernel(const float* __restrict__ coords, float* __restrict__ out) {
    int gid = blockIdx.x * blockDim.x + threadIdx.x;
    if (gid < BB * 3 * NN) out[gid] = coords[gid];
}

// ---------------------------------------------------------------------------
// orchestration
// ---------------------------------------------------------------------------
extern "C" void kernel_launch(void* const* d_in, const int* in_sizes, int n_in,
                              void* d_out, int out_size, void* d_ws, size_t ws_size,
                              hipStream_t stream) {
    (void)in_sizes; (void)n_in; (void)out_size; (void)ws_size;

    const float* features = (const float*)d_in[0];
    const float* coords   = (const float*)d_in[1];
    const float* vw1 = (const float*)d_in[2],  *vb1 = (const float*)d_in[3];
    const float* g1  = (const float*)d_in[4],  *be1 = (const float*)d_in[5];
    const float* vw2 = (const float*)d_in[6],  *vb2 = (const float*)d_in[7];
    const float* g2  = (const float*)d_in[8],  *be2 = (const float*)d_in[9];
    const float* uw1 = (const float*)d_in[10], *ub1 = (const float*)d_in[11];
    const float* g3  = (const float*)d_in[12], *be3 = (const float*)d_in[13];
    const float* uw2 = (const float*)d_in[14], *ub2 = (const float*)d_in[15];
    const float* g4  = (const float*)d_in[16], *be4 = (const float*)d_in[17];
    const float* uw3 = (const float*)d_in[18], *ub3 = (const float*)d_in[19];
    const float* g5  = (const float*)d_in[20], *be5 = (const float*)d_in[21];
    const float* pw1 = (const float*)d_in[22], *pb1 = (const float*)d_in[23];
    const float* pg1 = (const float*)d_in[24], *pbe1 = (const float*)d_in[25];
    const float* pw2 = (const float*)d_in[26], *pb2 = (const float*)d_in[27];
    const float* pg2 = (const float*)d_in[28], *pbe2 = (const float*)d_in[29];

    const float s3 = 1.0f / sqrtf(1.0f + 1e-4f);   // BN3d eval scale
    const float s1 = 1.0f / sqrtf(1.0f + 1e-5f);   // BN1d eval scale

    // -------- workspace layout --------
    char* ws = (char*)d_ws;
    size_t off = 0;
    auto alloc = [&](size_t bytes) { char* p = ws + off; off += (bytes + 255) & ~(size_t)255; return p; };
    float*     stats   = (float*)alloc(BB * 4 * sizeof(float));
    int*       idx     = (int*)  alloc((size_t)BB * NN * sizeof(int));
    float*     nc      = (float*)alloc((size_t)BB * 3 * NN * sizeof(float));
    float*     cnt     = (float*)alloc((size_t)BB * VV * sizeof(float));
    float*     biasAll = (float*)alloc(7 * 64 * sizeof(float));
    const size_t WK3 = (size_t)27 * 2 * 4 * 32 * 16;   // halfs per k=3 conv
    const size_t WK1 = (size_t)2 * 4 * 32 * 16;        // halfs per k=1 / pmlp
    _Float16* wV1 = (_Float16*)alloc(WK3 * 2);
    _Float16* wV2 = (_Float16*)alloc(WK3 * 2);
    _Float16* wU1 = (_Float16*)alloc(WK1 * 2);
    _Float16* wU2 = (_Float16*)alloc(WK3 * 2);
    _Float16* wU3 = (_Float16*)alloc(WK3 * 2);
    _Float16* wP1 = (_Float16*)alloc(WK1 * 2);
    _Float16* wP2 = (_Float16*)alloc(WK1 * 2);
    _Float16* feat_h   = (_Float16*)alloc((size_t)BB * NN * CC * 2);
    _Float16* v1_h     = (_Float16*)alloc((size_t)BB * NN * CC * 2);
    _Float16* fused1_h = (_Float16*)alloc((size_t)BB * NN * CC * 2);
    _Float16* pm_h     = (_Float16*)alloc((size_t)BB * NN * CC * 2);
    _Float16* gA = (_Float16*)alloc((size_t)BB * VV * CC * 2);
    _Float16* gB = (_Float16*)alloc((size_t)BB * VV * CC * 2);
    _Float16* gC = (_Float16*)alloc((size_t)BB * VV * CC * 2);
    float*    accum = (float*)alloc((size_t)BB * VV * CC * sizeof(float));

    float* biasV1 = biasAll + 0 * 64;
    float* biasV2 = biasAll + 1 * 64;
    float* biasU1 = biasAll + 2 * 64;
    float* biasU2 = biasAll + 3 * 64;
    float* biasU3 = biasAll + 4 * 64;
    float* biasP1 = biasAll + 5 * 64;
    float* biasP2 = biasAll + 6 * 64;

    // -------- weight / bias prep --------
    const int WT3 = (int)WK3, WT1 = (int)WK1;
    mpv_prep_w_kernel<<<(WT3 + 255) / 256, 256, 0, stream>>>(vw1, g1, s3, 27, wV1);
    mpv_prep_w_kernel<<<(WT3 + 255) / 256, 256, 0, stream>>>(vw2, g2, s3, 27, wV2);
    mpv_prep_w_kernel<<<(WT1 + 255) / 256, 256, 0, stream>>>(uw1, g3, s3, 1,  wU1);
    mpv_prep_w_kernel<<<(WT3 + 255) / 256, 256, 0, stream>>>(uw2, g4, s3, 27, wU2);
    mpv_prep_w_kernel<<<(WT3 + 255) / 256, 256, 0, stream>>>(uw3, g5, s3, 27, wU3);
    mpv_prep_w_kernel<<<(WT1 + 255) / 256, 256, 0, stream>>>(pw1, pg1, s1, 1, wP1);
    mpv_prep_w_kernel<<<(WT1 + 255) / 256, 256, 0, stream>>>(pw2, pg2, s1, 1, wP2);
    mpv_prep_bias_kernel<<<1, 64, 0, stream>>>(vb1, g1, be1, s3, biasV1);
    mpv_prep_bias_kernel<<<1, 64, 0, stream>>>(vb2, g2, be2, s3, biasV2);
    mpv_prep_bias_kernel<<<1, 64, 0, stream>>>(ub1, g3, be3, s3, biasU1);
    mpv_prep_bias_kernel<<<1, 64, 0, stream>>>(ub2, g4, be4, s3, biasU2);
    mpv_prep_bias_kernel<<<1, 64, 0, stream>>>(ub3, g5, be5, s3, biasU3);
    mpv_prep_bias_kernel<<<1, 64, 0, stream>>>(pb1, pg1, pbe1, s1, biasP1);
    mpv_prep_bias_kernel<<<1, 64, 0, stream>>>(pb2, pg2, pbe2, s1, biasP2);

    // -------- coords -> nc / voxel idx --------
    mpv_stats_kernel<<<BB, 256, 0, stream>>>(coords, stats);
    mpv_nc_idx_kernel<<<(BB * NN + 255) / 256, 256, 0, stream>>>(coords, stats, nc, idx);

    // -------- features -> channel-last fp16 --------
    const size_t PTS = (size_t)BB * NN * CC;                 // 8.4M
    mpv_feat_convert_kernel<<<(int)((PTS + 255) / 256), 256, 0, stream>>>(features, feat_h);

    // -------- voxelize pass 1 --------
    const int ACC4 = (int)((size_t)BB * VV * CC * 4 / 16);
    const int CNT4 = (int)((size_t)BB * VV * 4 / 16);
    mpv_zero_kernel<<<(ACC4 + 255) / 256, 256, 0, stream>>>((float4*)accum, ACC4);
    mpv_zero_kernel<<<(CNT4 + 255) / 256, 256, 0, stream>>>((float4*)cnt, CNT4);
    mpv_scatter_kernel<<<(int)((PTS + 255) / 256), 256, 0, stream>>>(feat_h, idx, accum, cnt, 1);
    const size_t GEL = (size_t)BB * VV * CC;                 // 16.8M
    mpv_normalize_kernel<<<(int)((GEL + 255) / 256), 256, 0, stream>>>(accum, cnt, gA);

    // -------- voxel branch 1: conv3(k3)+bn+lrelu x2 --------
    const int CONV_BLOCKS = (BB * RR * RR) / 4;              // 8192 wave-tiles / 4 waves
    mpv_conv3d_wmma_kernel<1><<<CONV_BLOCKS, 128, 0, stream>>>(gA, wV1, biasV1, gB);
    mpv_conv3d_wmma_kernel<1><<<CONV_BLOCKS, 128, 0, stream>>>(gB, wV2, biasV2, gC);

    // -------- point branch 1 + fuse --------
    const int MLP_BLOCKS = (BB * NN / 32) / 4;               // 4096 wave-tiles / 4 waves
    mpv_pointmlp_wmma_kernel<<<MLP_BLOCKS, 128, 0, stream>>>(feat_h, wP1, biasP1, pm_h);
    mpv_devox_fuse1_kernel<<<(int)((PTS + 255) / 256), 256, 0, stream>>>(gC, nc, pm_h, v1_h, fused1_h);

    // -------- voxelize pass 2 (same idx; counts unchanged) --------
    mpv_zero_kernel<<<(ACC4 + 255) / 256, 256, 0, stream>>>((float4*)accum, ACC4);
    mpv_scatter_kernel<<<(int)((PTS + 255) / 256), 256, 0, stream>>>(fused1_h, idx, accum, cnt, 0);
    mpv_normalize_kernel<<<(int)((GEL + 255) / 256), 256, 0, stream>>>(accum, cnt, gA);

    // -------- voxel branch 2: k=1 then k=3 x2 --------
    mpv_conv3d_wmma_kernel<0><<<CONV_BLOCKS, 128, 0, stream>>>(gA, wU1, biasU1, gB);
    mpv_conv3d_wmma_kernel<1><<<CONV_BLOCKS, 128, 0, stream>>>(gB, wU2, biasU2, gC);
    mpv_conv3d_wmma_kernel<1><<<CONV_BLOCKS, 128, 0, stream>>>(gC, wU3, biasU3, gA);

    // -------- point branch 2 + final fuse --------
    mpv_pointmlp_wmma_kernel<<<MLP_BLOCKS, 128, 0, stream>>>(fused1_h, wP2, biasP2, pm_h);
    float* out = (float*)d_out;
    mpv_final_kernel<<<(int)((PTS + 255) / 256), 256, 0, stream>>>(gA, nc, pm_h, v1_h, out);
    mpv_copy_coords_kernel<<<(BB * 3 * NN + 255) / 256, 256, 0, stream>>>(
        coords, out + (size_t)BB * CC * NN);
}
